// GTransformerHetero_73272142069811
// MI455X (gfx1250) — matrix operand order repaired and checked by reading, not compiled, not run on hardware
//
#include <hip/hip_runtime.h>

// ---------------------------------------------------------------------------
// GTransformer hetero layer for MI455X (gfx1250, wave32, WMMA, async-LDS).
//   1. zero z / h_update accumulators
//   2. convert+transpose weights to f16 (B-fragment friendly layout)
//   3. node GEMMs q,k,v,r: one async-LDS A-tile, A-frags reused x4   [WMMA]
//   4. edge GEMM e = y@We+be: double-buffered async-LDS pipeline     [WMMA]
//   5. edge scores + exp + segment-sum z  (atomics, L2-resident)
//   6. weighted aggregation into h_update (atomics)
//   7. gate GEMM over concat[h, r, h-r] (K=384), LDS-staged          [WMMA]
//   8. sigmoid-gate + LayerNorm + LeakyReLU -> out
// ---------------------------------------------------------------------------

typedef __attribute__((ext_vector_type(16))) _Float16 v16h;
typedef __attribute__((ext_vector_type(8)))  float    v8f;
typedef __attribute__((ext_vector_type(4)))  _Float16 v4h;

#define DD   128
#define HH   8
#define HD   16
#define EITER 8   // edge row-tiles per workgroup (double-buffered)

// ---- WMMA wrapper ----------------------------------------------------------
__device__ __forceinline__ v8f wmma_f16(v16h a, v16h b, v8f c) {
  return __builtin_amdgcn_wmma_f32_16x16x32_f16(
      /*neg_a=*/false, a, /*neg_b=*/false, b,
      /*c_mod=*/(short)0, c, /*reuse_a=*/false, /*reuse_b=*/false);
}

// ---- async global -> LDS copy of one 16x128 f32 tile (8 KB) ----------------
// 256 threads x 16 B x 2 rounds. Tracked by ASYNCcnt (2 per wave).
__device__ __forceinline__ void async_copy_tile_16x128(
    const float* __restrict__ gsrc, float* lds_dst, int tid) {
  const unsigned lbase = (unsigned)(uintptr_t)lds_dst;
#pragma unroll
  for (int rr = 0; rr < 2; ++rr) {
    const int off = (rr * 256 + tid) * 16;   // byte offset in tile
    const float* g = (const float*)((const char*)gsrc + off);
    const unsigned l = lbase + (unsigned)off;
    asm volatile("global_load_async_to_lds_b128 %0, %1, off"
                 :: "v"(l), "v"(g) : "memory");
  }
}
__device__ __forceinline__ void wait_async_0() {
  asm volatile("s_wait_asynccnt 0x0" ::: "memory");
}
__device__ __forceinline__ void wait_async_2() {
  asm volatile("s_wait_asynccnt 0x2" ::: "memory");
}

// ---- A fragment from an f32 row-major tile (LDS or global), cvt to f16 -----
// ISA 16x32 f16 A layout: lanes 0-15 (M=lane): halfs0-7=K0..7, halfs8-15=K16..23
//                         lanes16-31 (M=lane-16): halfs0-7=K8..15, halfs8-15=K24..31
__device__ __forceinline__ v16h a_frag_f32(const float* __restrict__ A,
                                           int lda, int k0, int lane) {
  const int row = lane & 15;
  const int klo = k0 + ((lane & 16) ? 8 : 0);
  const float* p = A + (long)row * lda + klo;
  const float4 u0 = ((const float4*)p)[0];
  const float4 u1 = ((const float4*)p)[1];
  const float4 u2 = ((const float4*)(p + 16))[0];
  const float4 u3 = ((const float4*)(p + 16))[1];
  v16h a;
  a[0]=(_Float16)u0.x; a[1]=(_Float16)u0.y; a[2]=(_Float16)u0.z; a[3]=(_Float16)u0.w;
  a[4]=(_Float16)u1.x; a[5]=(_Float16)u1.y; a[6]=(_Float16)u1.z; a[7]=(_Float16)u1.w;
  a[8]=(_Float16)u2.x; a[9]=(_Float16)u2.y; a[10]=(_Float16)u2.z; a[11]=(_Float16)u2.w;
  a[12]=(_Float16)u3.x; a[13]=(_Float16)u3.y; a[14]=(_Float16)u3.z; a[15]=(_Float16)u3.w;
  return a;
}

// ---- B fragment from pre-transposed f16 weights Wt[n][k] -------------------
__device__ __forceinline__ v16h b_frag_f16(const _Float16* __restrict__ Bt,
                                           int ldk, int n0, int k0, int lane) {
  const int col = n0 + (lane & 15);
  const int kb  = k0 + ((lane & 16) ? 8 : 0);
  typedef __attribute__((ext_vector_type(8))) _Float16 v8h_;
  const v8h_ lo = *(const v8h_*)(Bt + (long)col * ldk + kb);
  const v8h_ hi = *(const v8h_*)(Bt + (long)col * ldk + kb + 16);
  return __builtin_shufflevector(lo, hi, 0,1,2,3,4,5,6,7,8,9,10,11,12,13,14,15);
}

// ---- C tile stores (layout: VGPR j -> row j / j+8; lane -> column) ---------
__device__ __forceinline__ void store_tile_f16(_Float16* __restrict__ O, int ldo,
                                               long row0, int n0, v8f c,
                                               const float* __restrict__ bias,
                                               int lane) {
  const int  col = n0 + (lane & 15);
  const long rb  = row0 + ((lane & 16) ? 8 : 0);
  const float bv = bias[col];
#pragma unroll
  for (int j = 0; j < 8; ++j)
    O[(rb + j) * ldo + col] = (_Float16)(c[j] + bv);
}

__device__ __forceinline__ void store_tile_f32(float* __restrict__ O, int ldo,
                                               long row0, int n0, v8f c,
                                               const float* __restrict__ bias,
                                               int lane) {
  const int  col = n0 + (lane & 15);
  const long rb  = row0 + ((lane & 16) ? 8 : 0);
  const float bv = bias[col];
#pragma unroll
  for (int j = 0; j < 8; ++j)
    O[(rb + j) * ldo + col] = c[j] + bv;
}

// ---------------------------------------------------------------------------
// Kernel 0: zero a float buffer
// ---------------------------------------------------------------------------
__global__ void k_zero(float* __restrict__ p, long n) {
  long i = (long)blockIdx.x * blockDim.x + threadIdx.x;
  long stride = (long)gridDim.x * blockDim.x;
  for (; i < n; i += stride) p[i] = 0.0f;
}

// ---------------------------------------------------------------------------
// Kernel 1: convert f32 weight W[K][N] (row-major) -> transposed f16 Wt[N][K]
// ---------------------------------------------------------------------------
__global__ void k_wt_cvt(const float* __restrict__ W, _Float16* __restrict__ Wt,
                         int K, int N) {
  int idx = blockIdx.x * blockDim.x + threadIdx.x;
  if (idx >= K * N) return;
  int k = idx / N, n = idx % N;
  Wt[(long)n * K + k] = (_Float16)W[idx];
}

// ---------------------------------------------------------------------------
// Kernel 2: node projections q,k,v (f16) + residual r (f32). K = 128.
// grid = N/16, block = 256. One async-LDS A tile; A-frags reused for all
// four weight matrices (wave w covers cols [16w,16w+16) of each output).
// ---------------------------------------------------------------------------
__global__ __launch_bounds__(256) void k_node_proj(
    const float* __restrict__ x,
    const _Float16* __restrict__ WtQ, const _Float16* __restrict__ WtK,
    const _Float16* __restrict__ WtV, const _Float16* __restrict__ WtR,
    const float* __restrict__ bq, const float* __restrict__ bk,
    const float* __restrict__ bv, const float* __restrict__ br,
    _Float16* __restrict__ qh, _Float16* __restrict__ kh,
    _Float16* __restrict__ vh, float* __restrict__ r) {
  __shared__ __align__(128) float xs[16 * DD];
  const int tid  = threadIdx.x;
  const int lane = tid & 31;
  const int n0   = (tid >> 5) * 16;
  const long row0 = (long)blockIdx.x * 16;

  async_copy_tile_16x128(x + row0 * DD, xs, tid);
  wait_async_0();
  __syncthreads();

  v16h afr[4];
#pragma unroll
  for (int kt = 0; kt < 4; ++kt) afr[kt] = a_frag_f32(xs, DD, kt * 32, lane);

  const _Float16* Wts[4] = {WtQ, WtK, WtV, WtR};
  const float*    bia[4] = {bq, bk, bv, br};
#pragma unroll
  for (int m = 0; m < 4; ++m) {
    v8f c = {};
#pragma unroll
    for (int kt = 0; kt < 4; ++kt)
      c = wmma_f16(afr[kt], b_frag_f16(Wts[m], DD, n0, kt * 32, lane), c);
    if (m == 0)      store_tile_f16(qh, DD, row0, n0, c, bia[m], lane);
    else if (m == 1) store_tile_f16(kh, DD, row0, n0, c, bia[m], lane);
    else if (m == 2) store_tile_f16(vh, DD, row0, n0, c, bia[m], lane);
    else             store_tile_f32(r,  DD, row0, n0, c, bia[m], lane);
  }
}

// ---------------------------------------------------------------------------
// Kernel 3: edge projection e = y @ We + be (f16 out). K = 128.
// Double-buffered async-LDS pipeline, EITER row tiles per workgroup.
// ---------------------------------------------------------------------------
__global__ __launch_bounds__(256) void k_edge_proj(
    const float* __restrict__ y, const _Float16* __restrict__ WtE,
    const float* __restrict__ be, _Float16* __restrict__ eh) {
  __shared__ __align__(128) float ys[2][16 * DD];
  const int tid  = threadIdx.x;
  const int lane = tid & 31;
  const int n0   = (tid >> 5) * 16;
  const long tile0 = (long)blockIdx.x * EITER;

  // prologue: tile 0 in flight
  async_copy_tile_16x128(y + tile0 * (16 * DD), ys[0], tid);

  for (int t = 0; t < EITER; ++t) {
    if (t + 1 < EITER) {
      async_copy_tile_16x128(y + (tile0 + t + 1) * (16 * DD), ys[(t + 1) & 1], tid);
      wait_async_2();          // tile t's 2 copies done; next 2 still in flight
    } else {
      wait_async_0();
    }
    __syncthreads();           // tile t visible to all waves

    const float* A = ys[t & 1];
    v8f c = {};
#pragma unroll
    for (int kt = 0; kt < 4; ++kt) {
      v16h a = a_frag_f32(A, DD, kt * 32, lane);
      v16h b = b_frag_f16(WtE, DD, n0, kt * 32, lane);
      c = wmma_f16(a, b, c);
    }
    store_tile_f16(eh, DD, (tile0 + t) * 16, n0, c, be, lane);
    __syncthreads();           // everyone done reading buf before it is refilled
  }
}

// ---------------------------------------------------------------------------
// Kernel 4: edge attention score + exp + segment-sum denominator z.
// One wave per edge; lane l handles head l>>2, dims (l&3)*4 .. +3.
// ---------------------------------------------------------------------------
__global__ __launch_bounds__(256) void k_edge_score(
    const _Float16* __restrict__ qh, const _Float16* __restrict__ kh,
    const _Float16* __restrict__ eh, const int* __restrict__ src,
    const int* __restrict__ dst, float* __restrict__ esc,
    float* __restrict__ z, long E) {
  const long gid  = (long)blockIdx.x * blockDim.x + threadIdx.x;
  const long edge = gid >> 5;
  if (edge >= E) return;
  const int lane = threadIdx.x & 31;
  const int head = lane >> 2, part = lane & 3;
  const int d0   = head * HD + part * 4;
  const long s = src[edge], d = dst[edge];

  const v4h qv = *(const v4h*)(qh + d * DD + d0);
  const v4h kv = *(const v4h*)(kh + s * DD + d0);
  const v4h ev = *(const v4h*)(eh + edge * DD + d0);

  float acc = 0.f;
#pragma unroll
  for (int i = 0; i < 4; ++i)
    acc += (float)qv[i] * ((float)kv[i] + (float)ev[i]);
  acc += __shfl_xor(acc, 1);
  acc += __shfl_xor(acc, 2);

  if (part == 0) {
    const float es = expf(acc);
    esc[edge * HH + head] = es;
    atomicAdd(&z[d * HH + head], es);
  }
}

// ---------------------------------------------------------------------------
// Kernel 5: weighted aggregation  h_update[dst] += w * (v[src] + e).
// ---------------------------------------------------------------------------
__global__ __launch_bounds__(256) void k_edge_aggr(
    const _Float16* __restrict__ vh, const _Float16* __restrict__ eh,
    const int* __restrict__ src, const int* __restrict__ dst,
    const float* __restrict__ esc, const float* __restrict__ z,
    float* __restrict__ hupd, long E) {
  const long gid  = (long)blockIdx.x * blockDim.x + threadIdx.x;
  const long edge = gid >> 5;
  if (edge >= E) return;
  const int lane = threadIdx.x & 31;
  const int head = lane >> 2, part = lane & 3;
  const int d0   = head * HD + part * 4;
  const long s = src[edge], d = dst[edge];

  const float w = 0.25f * esc[edge * HH + head] / (1e-8f + z[d * HH + head]);

  const v4h vv = *(const v4h*)(vh + s * DD + d0);
  const v4h ev = *(const v4h*)(eh + edge * DD + d0);
#pragma unroll
  for (int i = 0; i < 4; ++i)
    atomicAdd(&hupd[d * DD + d0 + i], w * ((float)vv[i] + (float)ev[i]));
}

// ---------------------------------------------------------------------------
// Kernel 6: gate GEMM over concat[h, r, h-r]  (K = 384), LDS-staged. [WMMA]
// ---------------------------------------------------------------------------
__device__ __forceinline__ void cat_fill8(_Float16* a8,
                                          const float* __restrict__ h,
                                          const float* __restrict__ r,
                                          int row, int kk) {
  const int seg = kk >> 7;          // 0: h, 1: r, 2: h - r
  const int off = kk & 127;
  const float* hp = h + row * DD + off;
  const float* rp = r + row * DD + off;
#pragma unroll
  for (int i = 0; i < 8; ++i) {
    float v = (seg == 0) ? hp[i] : (seg == 1) ? rp[i] : (hp[i] - rp[i]);
    a8[i] = (_Float16)v;
  }
}

__global__ __launch_bounds__(256) void k_gate_gemm(
    const float* __restrict__ hupd, const float* __restrict__ resi,
    const _Float16* __restrict__ WtG, const float* __restrict__ bg,
    float* __restrict__ gate) {
  __shared__ __align__(128) float hs[16 * DD];
  __shared__ __align__(128) float rs[16 * DD];
  const int tid  = threadIdx.x;
  const int lane = tid & 31;
  const int n0   = (tid >> 5) * 16;
  const long row0 = (long)blockIdx.x * 16;

  async_copy_tile_16x128(hupd + row0 * DD, hs, tid);
  async_copy_tile_16x128(resi + row0 * DD, rs, tid);
  wait_async_0();
  __syncthreads();

  const int row = lane & 15;   // local row within LDS tile
  v8f c = {};
#pragma unroll
  for (int kt = 0; kt < 12; ++kt) {
    const int k0  = kt * 32;
    const int klo = k0 + ((lane & 16) ? 8 : 0);
    _Float16 t[16];
    cat_fill8(t, hs, rs, row, klo);
    cat_fill8(t + 8, hs, rs, row, klo + 16);
    v16h a;
#pragma unroll
    for (int i = 0; i < 16; ++i) a[i] = t[i];
    v16h b = b_frag_f16(WtG, 3 * DD, n0, k0, lane);
    c = wmma_f16(a, b, c);
  }
  store_tile_f32(gate, DD, row0, n0, c, bg, lane);   // pre-sigmoid + bias
}

// ---------------------------------------------------------------------------
// Kernel 7: sigmoid gate + gated residual + LayerNorm + LeakyReLU.
// ---------------------------------------------------------------------------
__global__ __launch_bounds__(256) void k_gate_ln_out(
    const float* __restrict__ hupd, const float* __restrict__ r,
    const float* __restrict__ gate, const float* __restrict__ gamma,
    const float* __restrict__ beta, float* __restrict__ out, long N) {
  const long gid  = (long)blockIdx.x * blockDim.x + threadIdx.x;
  const long node = gid >> 5;
  if (node >= N) return;
  const int lane = threadIdx.x & 31;
  const int d0   = lane * 4;

  const float4 h4 = *(const float4*)(hupd + node * DD + d0);
  const float4 r4 = *(const float4*)(r + node * DD + d0);
  const float4 g4 = *(const float4*)(gate + node * DD + d0);

  float u[4];
  {
    const float hh[4] = {h4.x, h4.y, h4.z, h4.w};
    const float rr[4] = {r4.x, r4.y, r4.z, r4.w};
    const float gg[4] = {g4.x, g4.y, g4.z, g4.w};
#pragma unroll
    for (int i = 0; i < 4; ++i) {
      const float b = 1.0f / (1.0f + expf(-gg[i]));
      u[i] = hh[i] + b * (rr[i] - hh[i]);     // h - b*h + b*r
    }
  }

  float s = u[0] + u[1] + u[2] + u[3];
#pragma unroll
  for (int m = 16; m >= 1; m >>= 1) s += __shfl_xor(s, m);
  const float mu = s * (1.0f / (float)DD);

  float vs = 0.f;
#pragma unroll
  for (int i = 0; i < 4; ++i) { const float dv = u[i] - mu; vs += dv * dv; }
#pragma unroll
  for (int m = 16; m >= 1; m >>= 1) vs += __shfl_xor(vs, m);
  const float inv = rsqrtf(vs * (1.0f / (float)DD) + 1e-5f);

#pragma unroll
  for (int i = 0; i < 4; ++i) {
    float o = (u[i] - mu) * inv * gamma[d0 + i] + beta[d0 + i];
    out[node * DD + d0 + i] = (o > 0.f) ? o : 0.01f * o;
  }
}

// ---------------------------------------------------------------------------
// Host-side orchestration
// ---------------------------------------------------------------------------
extern "C" void kernel_launch(void* const* d_in, const int* in_sizes, int n_in,
                              void* d_out, int out_size, void* d_ws, size_t ws_size,
                              hipStream_t stream) {
  const float* x     = (const float*)d_in[0];
  const float* y     = (const float*)d_in[1];
  const int*   src   = (const int*)d_in[2];
  const int*   dst   = (const int*)d_in[3];
  const float* Wq    = (const float*)d_in[4];
  const float* bq    = (const float*)d_in[5];
  const float* Wk    = (const float*)d_in[6];
  const float* bk    = (const float*)d_in[7];
  const float* Wv    = (const float*)d_in[8];
  const float* bv    = (const float*)d_in[9];
  const float* We    = (const float*)d_in[10];
  const float* be    = (const float*)d_in[11];
  const float* Wr    = (const float*)d_in[12];
  const float* br    = (const float*)d_in[13];
  const float* Wg    = (const float*)d_in[14];
  const float* bg    = (const float*)d_in[15];
  const float* gamma = (const float*)d_in[16];
  const float* beta  = (const float*)d_in[17];
  float* out = (float*)d_out;

  const long N = (long)in_sizes[0] / DD;   // 50000  (= 3125 * 16)
  const long E = (long)in_sizes[1] / DD;   // 640000 (= 40000 * 16)

  // ---- workspace carve-out (256B aligned) ----
  char* w = (char*)d_ws;
  auto take = [&](size_t bytes) {
    char* p = w; w += (bytes + 255) & ~(size_t)255; return p;
  };
  _Float16* qh   = (_Float16*)take(N * DD * sizeof(_Float16));
  _Float16* kh   = (_Float16*)take(N * DD * sizeof(_Float16));
  _Float16* vh   = (_Float16*)take(N * DD * sizeof(_Float16));
  _Float16* eh   = (_Float16*)take(E * DD * sizeof(_Float16));
  float*    resi = (float*)take(N * DD * sizeof(float));
  float*    hupd = (float*)take(N * DD * sizeof(float));
  float*    z    = (float*)take(N * HH * sizeof(float));
  float*    esc  = (float*)take(E * HH * sizeof(float));
  float*    gate = (float*)take(N * DD * sizeof(float));
  _Float16* WtQ  = (_Float16*)take(DD * DD * sizeof(_Float16));
  _Float16* WtK  = (_Float16*)take(DD * DD * sizeof(_Float16));
  _Float16* WtV  = (_Float16*)take(DD * DD * sizeof(_Float16));
  _Float16* WtR  = (_Float16*)take(DD * DD * sizeof(_Float16));
  _Float16* WtE  = (_Float16*)take(DD * DD * sizeof(_Float16));
  _Float16* WtG  = (_Float16*)take(3 * DD * DD * sizeof(_Float16));
  (void)ws_size; (void)n_in; (void)out_size;

  // ---- 0: zero accumulators ----
  k_zero<<<2048, 256, 0, stream>>>(hupd, N * DD);
  k_zero<<<512, 256, 0, stream>>>(z, N * HH);

  // ---- 1: weight convert/transpose ----
  const int wt128 = (DD * DD + 255) / 256;
  k_wt_cvt<<<wt128, 256, 0, stream>>>(Wq, WtQ, DD, DD);
  k_wt_cvt<<<wt128, 256, 0, stream>>>(Wk, WtK, DD, DD);
  k_wt_cvt<<<wt128, 256, 0, stream>>>(Wv, WtV, DD, DD);
  k_wt_cvt<<<wt128, 256, 0, stream>>>(Wr, WtR, DD, DD);
  k_wt_cvt<<<wt128, 256, 0, stream>>>(We, WtE, DD, DD);
  k_wt_cvt<<<(3 * DD * DD + 255) / 256, 256, 0, stream>>>(Wg, WtG, 3 * DD, DD);

  // ---- 2: node projections (q,k,v,r) ----
  k_node_proj<<<(unsigned)(N / 16), 256, 0, stream>>>(
      x, WtQ, WtK, WtV, WtR, bq, bk, bv, br, qh, kh, vh, resi);

  // ---- 3: edge projection e (double-buffered async pipeline) ----
  k_edge_proj<<<(unsigned)(E / 16 / EITER), 256, 0, stream>>>(y, WtE, be, eh);

  // ---- 4: scores + softmax denominator ----
  k_edge_score<<<(unsigned)((E * 32 + 255) / 256), 256, 0, stream>>>(
      qh, kh, eh, src, dst, esc, z, E);

  // ---- 5: aggregation ----
  k_edge_aggr<<<(unsigned)((E * 32 + 255) / 256), 256, 0, stream>>>(
      vh, eh, src, dst, esc, z, hupd, E);

  // ---- 6: gate GEMM (K=384) ----
  k_gate_gemm<<<(unsigned)(N / 16), 256, 0, stream>>>(hupd, resi, WtG, bg, gate);

  // ---- 7: gate + LayerNorm + LeakyReLU ----
  k_gate_ln_out<<<(unsigned)((N * 32 + 255) / 256), 256, 0, stream>>>(
      hupd, resi, gate, gamma, beta, out, N);
}
